// Bert_Lstm_CRF_48619029791211
// MI455X (gfx1250) — matrix operand order repaired and checked by reading, not compiled, not run on hardware
//
#include <hip/hip_runtime.h>
#include <hip/hip_bf16.h>
#include <math.h>

// ---------------- problem constants ----------------
#define BB   8
#define TT   256
#define HH   768
#define LL   12
#define NHH  12
#define DHH  64
#define FFF  3072
#define LHH  256
#define KLAB 9

// ---------------- CDNA5 async-to-LDS availability probe ----------------
#if defined(__has_builtin)
#if __has_builtin(__builtin_amdgcn_global_load_async_to_lds_b128) && \
    __has_builtin(__builtin_amdgcn_s_wait_asynccnt)
#define HAVE_ASYNC_LDS 1
#endif
#endif

typedef __attribute__((vector_size(16))) int vint4;
#define AS1 __attribute__((address_space(1)))
#define AS3 __attribute__((address_space(3)))

// ---------------- WMMA types ----------------
typedef __attribute__((ext_vector_type(16))) __bf16 v16bf;
typedef __attribute__((ext_vector_type(8)))  float  v8f;

union FragAB { uint4 u[2]; v16bf v; };

__device__ __forceinline__ unsigned short f2bf(float f) {
    unsigned int u = __float_as_uint(f);
    unsigned int r = u + 0x7FFFu + ((u >> 16) & 1u);   // round-to-nearest-even
    return (unsigned short)(r >> 16);
}
__device__ __forceinline__ float sigm(float x) { return 1.0f / (1.0f + __expf(-x)); }

// ================================================================
// bf16-WMMA GEMM, bf16 operands in global memory:
//   C[z] (f32 and/or bf16) = A[z] (MxK bf16, row major, lda)
//                          * B[z] (KxN, stored N-major: B(k,n)=Bm[n*ldb+k])
// BM=128, BN=64, BK=64, 256 threads = 8 waves, each wave 2x2 16x16 tiles,
// 8 v_wmma_f32_16x16x32_bf16 per stage.  Staging uses gfx1250
// global_load_async_to_lds_b128 when available (else plain b128 copies).
// Preconditions: M%128==0, K%64==0, lda%8==0, ldb%8==0 (all call sites OK);
// N guarded (classifier N=9).
// ================================================================
#define GBM 128
#define GBN 64
#define GBK 64
#define LDSK (GBK + 8)     // 72 ushorts -> 144B rows (16B aligned)

__global__ __launch_bounds__(256)
void gemm_bf16_wmma(const unsigned short* __restrict__ A,
                    const unsigned short* __restrict__ Bm,
                    float* __restrict__ Cf, unsigned short* __restrict__ C16,
                    int M, int N, int Kd, int lda, int ldb, int ldc,
                    long long sA, long long sB, long long sC,
                    const float* __restrict__ bias, int act)
{
    __shared__ __align__(16) unsigned short Als[GBM][LDSK];
    __shared__ __align__(16) unsigned short Btl[GBN][LDSK];

    const int z  = blockIdx.z;
    const unsigned short* Ab = A  + (size_t)z * sA;
    const unsigned short* Bb = Bm + (size_t)z * sB;

    const int m0  = blockIdx.y * GBM;
    const int n0  = blockIdx.x * GBN;
    const int tid = threadIdx.x;
    const int w    = tid >> 5;
    const int lane = tid & 31;
    const int waveM = (w & 3) * 32;
    const int waveN = (w >> 2) * 32;
    const int row16 = lane & 15;
    const int hi    = lane >> 4;

    // A staging: thread copies 32 contiguous K (4x b128) of one row
    const int ar = tid >> 1;              // 0..127
    const int ac = (tid & 1) * 32;        // 0 or 32
    const unsigned short* Aptr = Ab + (size_t)(m0 + ar) * lda + ac;
    // B staging: thread copies 16 contiguous K (2x b128) of one n-row
    const int bn  = tid >> 2;             // 0..63
    const int bk  = (tid & 3) * 16;       // 0,16,32,48
    const unsigned short* Bptr = Bb + (size_t)(n0 + bn) * ldb + bk;
    const bool bValid = (n0 + bn) < N;

    v8f acc[2][2] = {};

    for (int k0 = 0; k0 < Kd; k0 += GBK) {
        // ---------- stage A tile (128 x 64 bf16) ----------
        {
            const uint4* ap = (const uint4*)(Aptr + k0);
#ifdef HAVE_ASYNC_LDS
            __builtin_amdgcn_global_load_async_to_lds_b128(
                (AS1 vint4*)(ap), (AS3 vint4*)&Als[ar][ac], 0, 0);
            __builtin_amdgcn_global_load_async_to_lds_b128(
                (AS1 vint4*)(ap + 1), (AS3 vint4*)&Als[ar][ac + 8], 0, 0);
            __builtin_amdgcn_global_load_async_to_lds_b128(
                (AS1 vint4*)(ap + 2), (AS3 vint4*)&Als[ar][ac + 16], 0, 0);
            __builtin_amdgcn_global_load_async_to_lds_b128(
                (AS1 vint4*)(ap + 3), (AS3 vint4*)&Als[ar][ac + 24], 0, 0);
#else
            *(uint4*)&Als[ar][ac]     = ap[0];
            *(uint4*)&Als[ar][ac + 8] = ap[1];
            *(uint4*)&Als[ar][ac + 16] = ap[2];
            *(uint4*)&Als[ar][ac + 24] = ap[3];
#endif
            if (k0 + GBK < Kd)
                __builtin_prefetch(Aptr + k0 + GBK, 0, 1);   // global_prefetch_b8
        }
        // ---------- stage B tile (64 n-rows x 64 K bf16), N-guarded ----------
        if (bValid) {
            const uint4* bp = (const uint4*)(Bptr + k0);
#ifdef HAVE_ASYNC_LDS
            __builtin_amdgcn_global_load_async_to_lds_b128(
                (AS1 vint4*)(bp), (AS3 vint4*)&Btl[bn][bk], 0, 0);
            __builtin_amdgcn_global_load_async_to_lds_b128(
                (AS1 vint4*)(bp + 1), (AS3 vint4*)&Btl[bn][bk + 8], 0, 0);
#else
            *(uint4*)&Btl[bn][bk]     = bp[0];
            *(uint4*)&Btl[bn][bk + 8] = bp[1];
#endif
        } else {
            uint4 zz; zz.x = 0; zz.y = 0; zz.z = 0; zz.w = 0;
            *(uint4*)&Btl[bn][bk]     = zz;
            *(uint4*)&Btl[bn][bk + 8] = zz;
        }
#ifdef HAVE_ASYNC_LDS
        __builtin_amdgcn_s_wait_asynccnt(0);
#endif
        __syncthreads();

        // ---------- 2 K-steps of 32, 4 WMMA each ----------
#pragma unroll
        for (int kk = 0; kk < GBK; kk += 32) {
            FragAB af[2], bfr[2];
#pragma unroll
            for (int tmi = 0; tmi < 2; tmi++) {
                int rr = waveM + tmi * 16 + row16;
                af[tmi].u[0] = *(const uint4*)&Als[rr][kk + hi * 8];
                af[tmi].u[1] = *(const uint4*)&Als[rr][kk + hi * 8 + 16];
            }
#pragma unroll
            for (int tni = 0; tni < 2; tni++) {
                int nn = waveN + tni * 16 + row16;
                bfr[tni].u[0] = *(const uint4*)&Btl[nn][kk + hi * 8];
                bfr[tni].u[1] = *(const uint4*)&Btl[nn][kk + hi * 8 + 16];
            }
#pragma unroll
            for (int tmi = 0; tmi < 2; tmi++)
#pragma unroll
                for (int tni = 0; tni < 2; tni++)
                    acc[tmi][tni] = __builtin_amdgcn_wmma_f32_16x16x32_bf16(
                        false, af[tmi].v, false, bfr[tni].v,
                        (short)0, acc[tmi][tni], false, false);
        }
        __syncthreads();
    }

    // ---------- epilogue: bias / gelu / store f32 and/or bf16 ----------
#pragma unroll
    for (int tmi = 0; tmi < 2; tmi++)
#pragma unroll
        for (int tni = 0; tni < 2; tni++)
#pragma unroll
            for (int j = 0; j < 8; j++) {
                int gm = m0 + waveM + tmi * 16 + hi * 8 + j;
                int gn = n0 + waveN + tni * 16 + row16;
                if (gn < N) {
                    float v = acc[tmi][tni][j];
                    if (bias) v += bias[gn];
                    if (act)  v = 0.5f * v * (1.0f + erff(v * 0.70710678118f));
                    size_t ci = (size_t)z * sC + (size_t)gm * ldc + gn;
                    if (Cf)  Cf[ci]  = v;
                    if (C16) C16[ci] = f2bf(v);
                }
            }
}

// ================================================================
// weight conversion:  in f32 [K][N]  ->  out bf16 [N][K]  (transpose)
// ================================================================
__global__ __launch_bounds__(256)
void convT_kernel(const float* __restrict__ in, unsigned short* __restrict__ out,
                  int Kd, int N)
{
    int idx = blockIdx.x * 256 + threadIdx.x;      // over N*Kd, out-coalesced
    if (idx >= N * Kd) return;
    int k = idx % Kd;
    int n = idx / Kd;
    out[idx] = f2bf(in[(size_t)k * N + n]);
}

// plain f32 -> bf16
__global__ __launch_bounds__(256)
void conv_kernel(const float* __restrict__ in, unsigned short* __restrict__ out, int n)
{
    int idx = blockIdx.x * 256 + threadIdx.x;
    if (idx < n) out[idx] = f2bf(in[idx]);
}

// ================================================================
// Embedding sum:  x[row] = word_emb[id] + pos_emb[t] + type_emb[0]
// ================================================================
__global__ __launch_bounds__(256)
void embed_kernel(const int* __restrict__ ids, const float* __restrict__ we,
                  const float* __restrict__ pe, const float* __restrict__ te,
                  float* __restrict__ out)
{
    int row = blockIdx.x;                // b*T + t
    int t   = row % TT;
    int id  = ids[row];
    for (int i = threadIdx.x; i < HH; i += 256)
        out[(size_t)row * HH + i] = we[(size_t)id * HH + i] + pe[(size_t)t * HH + i] + te[i];
}

// ================================================================
// LayerNorm:  out = LN(x + res) * g + b  (res nullable); bf16 mirror optional
// ================================================================
__global__ __launch_bounds__(256)
void ln_kernel(const float* __restrict__ x, const float* __restrict__ res,
               const float* __restrict__ g, const float* __restrict__ b,
               float* __restrict__ out, unsigned short* __restrict__ outb)
{
    int row = blockIdx.x;
    const float* xr = x + (size_t)row * HH;
    const float* rr = res ? res + (size_t)row * HH : nullptr;
    __shared__ float s1[256], s2[256];
    float a = 0.0f, aa = 0.0f;
    for (int i = threadIdx.x; i < HH; i += 256) {
        float v = xr[i] + (rr ? rr[i] : 0.0f);
        a += v; aa += v * v;
    }
    s1[threadIdx.x] = a; s2[threadIdx.x] = aa;
    __syncthreads();
    for (int off = 128; off; off >>= 1) {
        if (threadIdx.x < off) { s1[threadIdx.x] += s1[threadIdx.x + off];
                                 s2[threadIdx.x] += s2[threadIdx.x + off]; }
        __syncthreads();
    }
    float mean = s1[0] / (float)HH;
    float var  = s2[0] / (float)HH - mean * mean;
    float rstd = rsqrtf(var + 1e-12f);
    for (int i = threadIdx.x; i < HH; i += 256) {
        float v = xr[i] + (rr ? rr[i] : 0.0f);
        float o = (v - mean) * rstd * g[i] + b[i];
        if (out)  out[(size_t)row * HH + i]  = o;
        if (outb) outb[(size_t)row * HH + i] = f2bf(o);
    }
}

// ================================================================
// bf16 layout shuffles for attention
// ================================================================
// [B*T, H] -> [B, NH, T, DH]
__global__ __launch_bounds__(256)
void permute_qk(const unsigned short* __restrict__ in, unsigned short* __restrict__ out)
{
    int idx = blockIdx.x * 256 + threadIdx.x;
    if (idx >= BB * TT * HH) return;
    int dh = idx % DHH;
    int h  = (idx / DHH) % NHH;
    int t  = (idx / HH) % TT;
    int b  = idx / (TT * HH);
    out[((((size_t)b * NHH + h) * TT + t) * DHH) + dh] = in[idx];
}
// [B*T, H] -> [B, NH, DH, T]   (V transposed so ctx GEMM B is N-major)
__global__ __launch_bounds__(256)
void permute_v(const unsigned short* __restrict__ in, unsigned short* __restrict__ out)
{
    int idx = blockIdx.x * 256 + threadIdx.x;
    if (idx >= BB * TT * HH) return;
    int dh = idx % DHH;
    int h  = (idx / DHH) % NHH;
    int t  = (idx / HH) % TT;
    int b  = idx / (TT * HH);
    out[((((size_t)b * NHH + h) * DHH + dh) * TT) + t] = in[idx];
}
// [B, NH, T, DH] -> [B*T, H]
__global__ __launch_bounds__(256)
void unpermute_ctx(const unsigned short* __restrict__ in, unsigned short* __restrict__ out)
{
    int idx = blockIdx.x * 256 + threadIdx.x;
    if (idx >= BB * TT * HH) return;
    int dh = idx % DHH;
    int h  = (idx / DHH) % NHH;
    int t  = (idx / HH) % TT;
    int b  = idx / (TT * HH);
    out[idx] = in[((((size_t)b * NHH + h) * TT + t) * DHH) + dh];
}

// ================================================================
// Row softmax (scale + mask bias), f32 scores in -> bf16 probs out.
// One block (256 threads) per row of [B*NH*T, T].
// ================================================================
__global__ __launch_bounds__(256)
void softmax_kernel(const float* __restrict__ s, unsigned short* __restrict__ pout,
                    const int* __restrict__ mask, float scale)
{
    int r = blockIdx.x;                  // over B*NH*T
    int b = r / (NHH * TT);
    const float* row = s + (size_t)r * TT;
    int k = threadIdx.x;
    float bias = (1.0f - (float)mask[b * TT + k]) * -10000.0f;
    float v = row[k] * scale + bias;
    __shared__ float red[256];
    red[k] = v; __syncthreads();
    for (int off = 128; off; off >>= 1) {
        if (k < off) red[k] = fmaxf(red[k], red[k + off]);
        __syncthreads();
    }
    float mx = red[0]; __syncthreads();
    float e = __expf(v - mx);
    red[k] = e; __syncthreads();
    for (int off = 128; off; off >>= 1) {
        if (k < off) red[k] += red[k + off];
        __syncthreads();
    }
    pout[(size_t)r * TT + k] = f2bf(e / red[0]);
}

// ================================================================
// BiLSTM recurrence. Block 0 = forward, block 1 = backward.
// 1024 threads; h, c, gates in LDS. Output bf16 into hcat[B*T][512].
// ================================================================
__global__ __launch_bounds__(1024)
void lstm_kernel(const float* __restrict__ xgf, const float* __restrict__ xgb,
                 const float* __restrict__ whhf, const float* __restrict__ bhhf,
                 const float* __restrict__ whhb, const float* __restrict__ bhhb,
                 unsigned short* __restrict__ hcat)
{
    const bool rev = (blockIdx.x == 1);
    const float* xg  = rev ? xgb  : xgf;
    const float* whh = rev ? whhb : whhf;
    const float* bhh = rev ? bhhb : bhhf;

    __shared__ float h[BB][LHH];
    __shared__ float c[BB][LHH];
    __shared__ float gates[4][BB][LHH];

    const int tid  = threadIdx.x;          // 0..1023
    const int grp  = tid >> 8;             // 0..3 (i,f,g,o)
    const int unit = tid & 255;

    for (int i = tid; i < BB * LHH; i += 1024) {
        ((float*)h)[i] = 0.0f;
        ((float*)c)[i] = 0.0f;
    }
    __syncthreads();

    const float bgate = bhh[tid];
    for (int step = 0; step < TT; step++) {
        const int t = rev ? (TT - 1 - step) : step;
        float accg[BB];
#pragma unroll
        for (int b = 0; b < BB; b++)
            accg[b] = bgate + xg[((size_t)b * TT + t) * (4 * LHH) + tid];
        for (int k2 = 0; k2 < LHH; k2++) {
            float wv = whh[(size_t)tid * LHH + k2];
#pragma unroll
            for (int b = 0; b < BB; b++)
                accg[b] += wv * h[b][k2];
        }
#pragma unroll
        for (int b = 0; b < BB; b++)
            gates[grp][b][unit] = accg[b];
        __syncthreads();

        int bq = tid >> 8;
#pragma unroll
        for (int rep = 0; rep < 2; rep++) {
            int b = bq + rep * 4;
            float ig = gates[0][b][unit];
            float fg = gates[1][b][unit];
            float gg = gates[2][b][unit];
            float og = gates[3][b][unit];
            float cn = sigm(fg) * c[b][unit] + sigm(ig) * tanhf(gg);
            float hn = sigm(og) * tanhf(cn);
            c[b][unit] = cn;
            h[b][unit] = hn;
            hcat[((size_t)b * TT + t) * (2 * LHH) + (rev ? LHH : 0) + unit] = f2bf(hn);
        }
        __syncthreads();
    }
}

// ================================================================
// CRF negative log-likelihood (gold score - logZ), averaged, negated.
// ================================================================
__global__ void crf_kernel(const float* __restrict__ em, const int* __restrict__ labels,
                           const int* __restrict__ mask,
                           const float* __restrict__ start, const float* __restrict__ endv,
                           const float* __restrict__ trans, float* __restrict__ out)
{
    __shared__ float part[BB];
    int b = threadIdx.x;
    if (b < BB) {
        const float* e  = em + (size_t)b * TT * KLAB;
        const int*   lb = labels + (size_t)b * TT;
        const int*   mk = mask   + (size_t)b * TT;

        int l0 = lb[0];
        float score = start[l0] + e[l0];
        int prev = l0;
        int msum = mk[0];
        for (int t = 1; t < TT; t++) {
            float m = (float)mk[t];
            int lt = lb[t];
            score += (trans[prev * KLAB + lt] + e[t * KLAB + lt]) * m;
            prev = lt;
            msum += mk[t];
        }
        int last = lb[msum - 1];
        score += endv[last];

        float alpha[KLAB];
        for (int j = 0; j < KLAB; j++) alpha[j] = start[j] + e[j];
        for (int t = 1; t < TT; t++) {
            float m = (float)mk[t];
            float nxt[KLAB];
            for (int j = 0; j < KLAB; j++) {
                float mx = -1e30f;
                for (int i = 0; i < KLAB; i++) {
                    float v = alpha[i] + trans[i * KLAB + j];
                    mx = fmaxf(mx, v);
                }
                float ss = 0.0f;
                for (int i = 0; i < KLAB; i++)
                    ss += __expf(alpha[i] + trans[i * KLAB + j] - mx);
                nxt[j] = mx + __logf(ss) + e[t * KLAB + j];
            }
            for (int j = 0; j < KLAB; j++)
                alpha[j] = (m > 0.0f) ? nxt[j] : alpha[j];
        }
        float mx = -1e30f;
        for (int j = 0; j < KLAB; j++) mx = fmaxf(mx, alpha[j] + endv[j]);
        float ss = 0.0f;
        for (int j = 0; j < KLAB; j++) ss += __expf(alpha[j] + endv[j] - mx);
        float logZ = mx + __logf(ss);
        part[b] = score - logZ;
    }
    __syncthreads();
    if (threadIdx.x == 0) {
        float s = 0.0f;
        for (int i = 0; i < BB; i++) s += part[i];
        out[0] = -s / (float)BB;
    }
}

// ================================================================
// host-side helpers
// ================================================================
static inline void launch_gemm(hipStream_t st,
                               const unsigned short* A, const unsigned short* Bm,
                               float* Cf, unsigned short* C16,
                               int M, int N, int Kd, int lda, int ldb, int ldc,
                               long long sA, long long sB, long long sC, int batch,
                               const float* bias, int act)
{
    dim3 grid((N + GBN - 1) / GBN, (M + GBM - 1) / GBM, batch);
    gemm_bf16_wmma<<<grid, 256, 0, st>>>(A, Bm, Cf, C16, M, N, Kd, lda, ldb, ldc,
                                         sA, sB, sC, bias, act);
}

extern "C" void kernel_launch(void* const* d_in, const int* in_sizes, int n_in,
                              void* d_out, int out_size, void* d_ws, size_t ws_size,
                              hipStream_t stream)
{
    (void)in_sizes; (void)n_in; (void)out_size; (void)ws_size;

    const int*   input_ids = (const int*)d_in[0];
    const int*   attn_mask = (const int*)d_in[1];
    const int*   labels    = (const int*)d_in[2];
    const float* word_emb  = (const float*)d_in[3];
    const float* pos_emb   = (const float*)d_in[4];
    const float* type_emb  = (const float*)d_in[5];
    const float* emb_ln_g  = (const float*)d_in[6];
    const float* emb_ln_b  = (const float*)d_in[7];
    const float* Wq = (const float*)d_in[8];   const float* bq = (const float*)d_in[9];
    const float* Wk = (const float*)d_in[10];  const float* bk = (const float*)d_in[11];
    const float* Wv = (const float*)d_in[12];  const float* bv = (const float*)d_in[13];
    const float* Wo = (const float*)d_in[14];  const float* bo = (const float*)d_in[15];
    const float* ln1_g = (const float*)d_in[16]; const float* ln1_b = (const float*)d_in[17];
    const float* Wf1 = (const float*)d_in[18];  const float* bf1 = (const float*)d_in[19];
    const float* Wf2 = (const float*)d_in[20];  const float* bf2 = (const float*)d_in[21];
    const float* ln2_g = (const float*)d_in[22]; const float* ln2_b = (const float*)d_in[23];
    const float* w_ih_f = (const float*)d_in[24]; const float* w_hh_f = (const float*)d_in[25];
    const float* b_ih_f = (const float*)d_in[26]; const float* b_hh_f = (const float*)d_in[27];
    const float* w_ih_b = (const float*)d_in[28]; const float* w_hh_b = (const float*)d_in[29];
    const float* b_ih_b = (const float*)d_in[30]; const float* b_hh_b = (const float*)d_in[31];
    const float* W_clf = (const float*)d_in[32];  const float* b_clf = (const float*)d_in[33];
    const float* crf_start = (const float*)d_in[34];
    const float* crf_end   = (const float*)d_in[35];
    const float* crf_trans = (const float*)d_in[36];

    // ---------------- workspace layout ----------------
    char* ws = (char*)d_ws;
    size_t off = 0;
    auto wallocf = [&](size_t n) -> float* {                 // n floats
        float* p = (float*)(ws + off);
        off += ((n * 4 + 255) & ~(size_t)255);
        return p;
    };
    auto wallocb = [&](size_t n) -> unsigned short* {        // n bf16
        unsigned short* p = (unsigned short*)(ws + off);
        off += ((n * 2 + 255) & ~(size_t)255);
        return p;
    };
    const size_t SZ_BTH = (size_t)BB * TT * HH;              // 1.57M elems
    // f32 buffers
    float* x0  = wallocf(SZ_BTH);                            // layer state
    float* x1  = wallocf(SZ_BTH);                            // attn-LN out "a"
    float* t1  = wallocf(SZ_BTH);                            // projection out
    float* sc  = wallocf((size_t)BB * NHH * TT * TT);        // attn scores f32
    float* xgf = wallocf((size_t)BB * TT * 4 * LHH);
    float* xgb = wallocf((size_t)BB * TT * 4 * LHH);
    float* em  = wallocf((size_t)BB * TT * KLAB);
    // bf16 buffers
    unsigned short* x0b   = wallocb(SZ_BTH);
    unsigned short* x1b   = wallocb(SZ_BTH);
    unsigned short* qkraw = wallocb(SZ_BTH);                 // raw qkv / scratch
    unsigned short* qpb   = wallocb(SZ_BTH);                 // Q perm / ctx perm
    unsigned short* kpb   = wallocb(SZ_BTH);
    unsigned short* vptb  = wallocb(SZ_BTH);                 // V transposed perm
    unsigned short* ctxhb = wallocb(SZ_BTH);
    unsigned short* bigb  = wallocb((size_t)BB * NHH * TT * TT); // probs / FF1-out
    unsigned short* hcatb = wallocb((size_t)BB * TT * 2 * LHH);
    // per-layer transposed bf16 weights (reused each layer)
    unsigned short* wqT  = wallocb((size_t)HH * HH);
    unsigned short* wkT  = wallocb((size_t)HH * HH);
    unsigned short* wvT  = wallocb((size_t)HH * HH);
    unsigned short* woT  = wallocb((size_t)HH * HH);
    unsigned short* wf1T = wallocb((size_t)HH * FFF);
    unsigned short* wf2T = wallocb((size_t)HH * FFF);
    // one-time bf16 weights
    unsigned short* wihfB = wallocb((size_t)4 * LHH * HH);   // already [N][K]
    unsigned short* wihbB = wallocb((size_t)4 * LHH * HH);
    unsigned short* wclfT = wallocb((size_t)KLAB * 2 * LHH);

    const int ROWS = BB * TT;                                // 2048
    const int permBlocks = (BB * TT * HH + 255) / 256;
    auto cdiv = [](size_t n, size_t d) { return (int)((n + d - 1) / d); };

    // ---------------- one-time weight conversions ----------------
    conv_kernel<<<cdiv((size_t)4 * LHH * HH, 256), 256, 0, stream>>>(w_ih_f, wihfB, 4 * LHH * HH);
    conv_kernel<<<cdiv((size_t)4 * LHH * HH, 256), 256, 0, stream>>>(w_ih_b, wihbB, 4 * LHH * HH);
    convT_kernel<<<cdiv((size_t)KLAB * 2 * LHH, 256), 256, 0, stream>>>(W_clf, wclfT, 2 * LHH, KLAB);

    // ---------------- embeddings ----------------
    embed_kernel<<<ROWS, 256, 0, stream>>>(input_ids, word_emb, pos_emb, type_emb, x1);
    ln_kernel<<<ROWS, 256, 0, stream>>>(x1, nullptr, emb_ln_g, emb_ln_b, x0, x0b);

    // ---------------- transformer layers ----------------
    const int nHH  = cdiv((size_t)HH * HH, 256);
    const int nHFF = cdiv((size_t)HH * FFF, 256);
    for (int l = 0; l < LL; l++) {
        // transpose+convert this layer's weights to N-major bf16
        convT_kernel<<<nHH, 256, 0, stream>>>(Wq + (size_t)l * HH * HH, wqT, HH, HH);
        convT_kernel<<<nHH, 256, 0, stream>>>(Wk + (size_t)l * HH * HH, wkT, HH, HH);
        convT_kernel<<<nHH, 256, 0, stream>>>(Wv + (size_t)l * HH * HH, wvT, HH, HH);
        convT_kernel<<<nHH, 256, 0, stream>>>(Wo + (size_t)l * HH * HH, woT, HH, HH);
        convT_kernel<<<nHFF, 256, 0, stream>>>(Wf1 + (size_t)l * HH * FFF, wf1T, HH, FFF);
        convT_kernel<<<nHFF, 256, 0, stream>>>(Wf2 + (size_t)l * HH * FFF, wf2T, FFF, HH);

        // Q,K,V projections (+bias) -> bf16, then permute
        launch_gemm(stream, x0b, wqT, nullptr, qkraw, ROWS, HH, HH, HH, HH, HH,
                    0, 0, 0, 1, bq + (size_t)l * HH, 0);
        permute_qk<<<permBlocks, 256, 0, stream>>>(qkraw, qpb);
        launch_gemm(stream, x0b, wkT, nullptr, qkraw, ROWS, HH, HH, HH, HH, HH,
                    0, 0, 0, 1, bk + (size_t)l * HH, 0);
        permute_qk<<<permBlocks, 256, 0, stream>>>(qkraw, kpb);
        launch_gemm(stream, x0b, wvT, nullptr, qkraw, ROWS, HH, HH, HH, HH, HH,
                    0, 0, 0, 1, bv + (size_t)l * HH, 0);
        permute_v<<<permBlocks, 256, 0, stream>>>(qkraw, vptb);

        // scores = Q @ K^T (batched over B*NH = 96), f32 out
        launch_gemm(stream, qpb, kpb, sc, nullptr, TT, TT, DHH, DHH, DHH, TT,
                    (long long)TT * DHH, (long long)TT * DHH, (long long)TT * TT,
                    BB * NHH, nullptr, 0);
        // softmax -> bf16 probs
        softmax_kernel<<<BB * NHH * TT, 256, 0, stream>>>(sc, bigb, attn_mask, 0.125f);
        // ctx = P @ V (B = V transposed, N-major), bf16 out into qpb
        launch_gemm(stream, bigb, vptb, nullptr, qpb, TT, DHH, TT, TT, TT, DHH,
                    (long long)TT * TT, (long long)TT * DHH, (long long)TT * DHH,
                    BB * NHH, nullptr, 0);
        unpermute_ctx<<<permBlocks, 256, 0, stream>>>(qpb, ctxhb);

        // output projection + residual LN
        launch_gemm(stream, ctxhb, woT, t1, nullptr, ROWS, HH, HH, HH, HH, HH,
                    0, 0, 0, 1, bo + (size_t)l * HH, 0);
        ln_kernel<<<ROWS, 256, 0, stream>>>(t1, x0, ln1_g + (size_t)l * HH,
                                            ln1_b + (size_t)l * HH, x1, x1b);

        // FFN: gelu(a@Wf1+bf1) (bf16) @Wf2+bf2, residual LN back into x0
        launch_gemm(stream, x1b, wf1T, nullptr, bigb, ROWS, FFF, HH, HH, HH, FFF,
                    0, 0, 0, 1, bf1 + (size_t)l * FFF, 1 /*gelu*/);
        launch_gemm(stream, bigb, wf2T, t1, nullptr, ROWS, HH, FFF, FFF, FFF, HH,
                    0, 0, 0, 1, bf2 + (size_t)l * HH, 0);
        ln_kernel<<<ROWS, 256, 0, stream>>>(t1, x1, ln2_g + (size_t)l * HH,
                                            ln2_b + (size_t)l * HH, x0, x0b);
    }

    // ---------------- BiLSTM ----------------
    launch_gemm(stream, x0b, wihfB, xgf, nullptr, ROWS, 4 * LHH, HH, HH, HH, 4 * LHH,
                0, 0, 0, 1, b_ih_f, 0);
    launch_gemm(stream, x0b, wihbB, xgb, nullptr, ROWS, 4 * LHH, HH, HH, HH, 4 * LHH,
                0, 0, 0, 1, b_ih_b, 0);
    lstm_kernel<<<2, 1024, 0, stream>>>(xgf, xgb, w_hh_f, b_hh_f, w_hh_b, b_hh_b, hcatb);

    // ---------------- classifier ----------------
    launch_gemm(stream, hcatb, wclfT, em, nullptr, ROWS, KLAB, 2 * LHH,
                2 * LHH, 2 * LHH, KLAB, 0, 0, 0, 1, b_clf, 0);

    // ---------------- CRF loss ----------------
    crf_kernel<<<1, 32, 0, stream>>>(em, labels, attn_mask,
                                     crf_start, crf_end, crf_trans, (float*)d_out);
}
